// MyModel_21577915695161
// MI455X (gfx1250) — matrix operand order repaired and checked by reading, not compiled, not run on hardware
//
#include <hip/hip_runtime.h>
#include <hip/hip_bf16.h>
#include <math.h>

// ---------------- problem constants ----------------
constexpr int B  = 64;
constexpr int T  = 512;
constexpr int E  = 256;
constexpr int U  = 1024;
constexpr int V  = 96;
constexpr int U3 = 3 * U;      // 3072
constexpr int BT = B * T;      // 32768

// ---------------- vector types ----------------
typedef __attribute__((ext_vector_type(16))) __bf16 v16bf;
typedef __attribute__((ext_vector_type(8)))  __bf16 v8bf;
typedef __attribute__((ext_vector_type(8)))  float  v8f;
typedef __attribute__((ext_vector_type(4)))  int    v4i;

// ---------------- async-to-LDS availability ----------------
#if defined(__has_builtin)
#if __has_builtin(__builtin_amdgcn_global_load_async_to_lds_b128)
#define HAVE_ASYNC_LDS 1
#endif
#endif
#ifndef HAVE_ASYNC_LDS
#define HAVE_ASYNC_LDS 0
#endif

#if HAVE_ASYNC_LDS
typedef __attribute__((address_space(1))) v4i* g_v4i_p;   // global int4*
typedef __attribute__((address_space(3))) v4i* l_v4i_p;   // LDS int4*
#endif

template <int N>
__device__ __forceinline__ void wait_async() {
#if HAVE_ASYNC_LDS
#if __has_builtin(__builtin_amdgcn_s_wait_asynccnt)
    __builtin_amdgcn_s_wait_asynccnt(N);
#else
    if (N == 0) asm volatile("s_wait_asynccnt 0x0" ::: "memory");
    else        asm volatile("s_wait_asynccnt 0x4" ::: "memory");
#endif
#endif
}

// copy one 16B packet into LDS (async when available)
__device__ __forceinline__ void copy16_to_lds(__bf16* l, const __bf16* g) {
#if HAVE_ASYNC_LDS
    __builtin_amdgcn_global_load_async_to_lds_b128((g_v4i_p)(void*)(__bf16*)g,
                                                   (l_v4i_p)(void*)l, 0, 0);
#else
    *(v8bf*)l = *(const v8bf*)g;
#endif
}

// ---------------- WMMA fragment helpers ----------------
// A fragment: 16x32 bf16 tile, row-major source. CDNA5 layout:
// lanes 0-15 hold rows M=0..15; lane-group (lane>>4) selects K-subgroup.
// Per lane: elements 0..7  -> K = grp*8 + 0..7      (contiguous)
//           elements 8..15 -> K = 16 + grp*8 + 0..7 (contiguous)
__device__ __forceinline__ v16bf load_frag_a(const __bf16* base, int row_stride) {
    const int lane = threadIdx.x & 31;
    const int row  = lane & 15;
    const int grp  = lane >> 4;
    const __bf16* p = base + (long)row * row_stride + grp * 8;
    v8bf lo = *(const v8bf*)(p);
    v8bf hi = *(const v8bf*)(p + 16);
    v16bf a;
#pragma unroll
    for (int i = 0; i < 8; ++i) { a[i] = lo[i]; a[i + 8] = hi[i]; }
    return a;
}

// B fragment: 32x16 bf16 tile, from a TRANSPOSED copy Bt[N][K]
// lanes 0-15 hold column n=lane; lane-group (lane>>4) selects K-subgroup.
// Per lane: elements 0..7 -> K = grp*8 + 0..7; elements 8..15 -> K = 16 + grp*8 + 0..7.
__device__ __forceinline__ v16bf load_frag_bt(const __bf16* base, int row_stride) {
    const int lane = threadIdx.x & 31;
    const int n    = lane & 15;
    const int grp  = lane >> 4;
    const __bf16* p = base + (long)n * row_stride + grp * 8;
    v8bf lo = *(const v8bf*)(p);
    v8bf hi = *(const v8bf*)(p + 16);
    v16bf b;
#pragma unroll
    for (int i = 0; i < 8; ++i) { b[i] = lo[i]; b[i + 8] = hi[i]; }
    return b;
}

__device__ __forceinline__ v8f wmma_bf16(v16bf a, v16bf b, v8f c) {
    return __builtin_amdgcn_wmma_f32_16x16x32_bf16(false, a, false, b, (short)0, c,
                                                   false, false);
}

// C/D 16x16 f32: lane holds column n=lane&15; VGPR r holds row m = r + 8*(lane>>4).
__device__ __forceinline__ void store_c_bias(float* out, int ldn, int m0, int n0,
                                             const float* bias, v8f c) {
    const int lane = threadIdx.x & 31;
    const int n    = n0 + (lane & 15);
    const int grp  = lane >> 4;
    const float bv = bias ? bias[n] : 0.f;
#pragma unroll
    for (int r = 0; r < 8; ++r)
        out[(size_t)(m0 + grp * 8 + r) * ldn + n] = c[r] + bv;
}

__device__ __forceinline__ float sigmoidf_(float x) {
    return 1.f / (1.f + __expf(-x));
}

// ---------------- one-time pack kernels ----------------
__global__ void k_trans_bf16(const float* __restrict__ src, __bf16* __restrict__ dst,
                             int K, int N) {
    int idx = blockIdx.x * blockDim.x + threadIdx.x;
    if (idx >= K * N) return;
    int n = idx / K;
    int k = idx - n * K;
    dst[idx] = (__bf16)src[(size_t)k * N + n];
}

__global__ void k_embed(const int* __restrict__ tokens, const float* __restrict__ emb,
                        __bf16* __restrict__ xbf) {
    int idx = blockIdx.x * blockDim.x + threadIdx.x;   // over BT * E/4
    int bt  = idx / (E / 4);
    int e4  = (idx - bt * (E / 4)) * 4;
    if (bt >= BT) return;
    int tok = tokens[bt];
    const float* s = emb + (size_t)tok * E + e4;
    __bf16* d = xbf + (size_t)bt * E + e4;
    d[0] = (__bf16)s[0]; d[1] = (__bf16)s[1];
    d[2] = (__bf16)s[2]; d[3] = (__bf16)s[3];
}

__global__ void k_inith(const float* __restrict__ h0f, const float* __restrict__ h0b,
                        __bf16* __restrict__ hbuf) {
    int idx = blockIdx.x * blockDim.x + threadIdx.x;
    if (idx >= 2 * B * U) return;
    float v = (idx < B * U) ? h0f[idx] : h0b[idx - B * U];
    hbuf[idx] = (__bf16)v;
}

// ---------------- input projection GEMM ----------------
// xp[dir][bt][3U] = x(_rev) @ W + b[0].   grid: (BT/64, U3/128, 2), block 256.
// A and B fragments register double-buffered (prefetch k+32 before k's WMMAs).
__global__ void k_xproj(const __bf16* __restrict__ xbf,
                        const __bf16* __restrict__ Wft, const __bf16* __restrict__ Wbt,
                        const float* __restrict__ bf0, const float* __restrict__ bb0,
                        float* __restrict__ xpF, float* __restrict__ xpB) {
    const int dir = blockIdx.z;
    const __bf16* Wt   = dir ? Wbt : Wft;
    const float*  bias = dir ? bb0 : bf0;
    float*        out  = dir ? xpB : xpF;

    const int wave  = threadIdx.x >> 5;
    const int m0    = blockIdx.x * 64 + (wave & 3) * 16;
    const int nbase = blockIdx.y * 128 + (wave >> 2) * 64;

    int rowStride = E;
    const __bf16* abase;
    if (dir == 0) {
        abase = xbf + (size_t)m0 * E;
    } else {
        int b  = m0 >> 9;        // / T
        int t0 = m0 & (T - 1);
        abase  = xbf + ((size_t)b * T + (T - 1 - t0)) * E;
        rowStride = -E;
    }
    const __bf16* w0 = Wt + (size_t)(nbase +  0) * E;
    const __bf16* w1 = Wt + (size_t)(nbase + 16) * E;
    const __bf16* w2 = Wt + (size_t)(nbase + 32) * E;
    const __bf16* w3 = Wt + (size_t)(nbase + 48) * E;

    v8f a0 = {}, a1 = {}, a2 = {}, a3 = {};
    // prime k = 0 fragments
    v16bf ac  = load_frag_a(abase, rowStride);
    v16bf b0c = load_frag_bt(w0, E);
    v16bf b1c = load_frag_bt(w1, E);
    v16bf b2c = load_frag_bt(w2, E);
    v16bf b3c = load_frag_bt(w3, E);
#pragma unroll
    for (int k = 0; k < E; k += 32) {
        v16bf an = ac, b0n = b0c, b1n = b1c, b2n = b2c, b3n = b3c;
        if (k + 32 < E) {
            an  = load_frag_a(abase + k + 32, rowStride);
            b0n = load_frag_bt(w0 + k + 32, E);
            b1n = load_frag_bt(w1 + k + 32, E);
            b2n = load_frag_bt(w2 + k + 32, E);
            b3n = load_frag_bt(w3 + k + 32, E);
        }
        a0 = wmma_bf16(ac, b0c, a0);
        a1 = wmma_bf16(ac, b1c, a1);
        a2 = wmma_bf16(ac, b2c, a2);
        a3 = wmma_bf16(ac, b3c, a3);
        ac = an; b0c = b0n; b1c = b1n; b2c = b2n; b3c = b3n;
    }
    store_c_bias(out, U3, m0, nbase +  0, bias, a0);
    store_c_bias(out, U3, m0, nbase + 16, bias, a1);
    store_c_bias(out, U3, m0, nbase + 32, bias, a2);
    store_c_bias(out, U3, m0, nbase + 48, bias, a3);
}

// ---------------- recurrent step (fused GEMM + gates) ----------------
// Workgroup: all M=64 batch rows x one 32-unit block, K=U reduction.
// h (A matrix) is staged through LDS in K-chunks of 128 via async copies,
// double-buffered; U-matrix B fragments are register double-buffered.
// grid: (U/32, 2 dirs), block 256 (8 waves).
__global__ void k_gru_step(int t,
                           const __bf16* __restrict__ hprev, __bf16* __restrict__ hnext,
                           const __bf16* __restrict__ Uft, const __bf16* __restrict__ Ubt,
                           const float* __restrict__ bfr, const float* __restrict__ bbr,
                           const float* __restrict__ xpF, const float* __restrict__ xpB,
                           __bf16* __restrict__ h_all, float* __restrict__ out_h) {
    constexpr int KC    = 128;       // K-chunk staged in LDS
    constexpr int PITCH = KC + 8;    // 272B row stride -> 4-bank rotation/row
    constexpr int NCH   = U / KC;    // 8 chunks
    __shared__ __bf16 sbuf[2][64 * PITCH];

    const int dir = blockIdx.y;
    const __bf16* Ut   = dir ? Ubt : Uft;
    const float*  brec = dir ? bbr : bfr;
    const float*  xp   = dir ? xpB : xpF;

    const int wave = threadIdx.x >> 5;
    const int m0   = (wave & 3) * 16;
    const int u0   = blockIdx.x * 32 + (wave >> 2) * 16;
    const __bf16* hp = hprev + (size_t)dir * (B * U);

    // stage chunk c of h into sbuf[c&1]: 64 rows x 128 cols, 4x b128 per thread
    auto issue_chunk = [&](int c) {
        const int tid = threadIdx.x;
#pragma unroll
        for (int j = 0; j < 4; ++j) {
            int idx = j * 256 + tid;
            int row = idx >> 4;
            int col = (idx & 15) * 8;
            copy16_to_lds(&sbuf[c & 1][row * PITCH + col],
                          hp + (size_t)row * U + (size_t)c * KC + col);
        }
    };

    v8f cz = {}, cr = {}, ch = {};
    // prime B fragments for k = 0
    v16bf bzc = load_frag_bt(Ut + (size_t)(u0)         * U, U);
    v16bf brc = load_frag_bt(Ut + (size_t)(U + u0)     * U, U);
    v16bf bhc = load_frag_bt(Ut + (size_t)(2 * U + u0) * U, U);
    issue_chunk(0);

    for (int c = 0; c < NCH; ++c) {
        if (c + 1 < NCH) { issue_chunk(c + 1); wait_async<4>(); }
        else             { wait_async<0>(); }
        __syncthreads();                       // chunk c resident for all waves
        const __bf16* abuf = &sbuf[c & 1][m0 * PITCH];
#pragma unroll
        for (int kk = 0; kk < KC; kk += 32) {
            const int k = c * KC + kk;
            v16bf a = load_frag_a(abuf + kk, PITCH);       // ds loads
            v16bf bzn = bzc, brn = brc, bhn = bhc;
            if (k + 32 < U) {                               // prefetch next B
                bzn = load_frag_bt(Ut + (size_t)(u0)         * U + k + 32, U);
                brn = load_frag_bt(Ut + (size_t)(U + u0)     * U + k + 32, U);
                bhn = load_frag_bt(Ut + (size_t)(2 * U + u0) * U + k + 32, U);
            }
            cz = wmma_bf16(a, bzc, cz);
            cr = wmma_bf16(a, brc, cr);
            ch = wmma_bf16(a, bhc, ch);
            bzc = bzn; brc = brn; bhc = bhn;
        }
        __syncthreads();                       // done reading before overwrite
    }

    const int lane = threadIdx.x & 31;
    const int u    = u0 + (lane & 15);
    const int grp  = lane >> 4;
    const float bz_ = brec[u];
    const float br_ = brec[U + u];
    const float bh_ = brec[2 * U + u];
    const int t_out = dir ? (T - 1 - t) : t;

#pragma unroll
    for (int r = 0; r < 8; ++r) {
        const int b = m0 + grp * 8 + r;
        const size_t xrow = ((size_t)b * T + t) * (size_t)U3;
        float xz = xp[xrow + u];
        float xr = xp[xrow + U + u];
        float xh = xp[xrow + 2 * U + u];
        float z  = sigmoidf_(xz + cz[r] + bz_);
        float rg = sigmoidf_(xr + cr[r] + br_);
        float hh = tanhf(xh + rg * (ch[r] + bh_));
        float hv = (float)hp[(size_t)b * U + u];
        float hn = z * hv + (1.f - z) * hh;
        hnext[((size_t)dir * B + b) * U + u] = (__bf16)hn;
        h_all[((size_t)b * T + t_out) * (2 * U) + (size_t)dir * U + u] = (__bf16)hn;
        if (t == T - 1)
            out_h[(size_t)dir * (B * U) + (size_t)b * U + u] = hn;   // hf / hb
    }
}

// ---------------- decoder GEMM ----------------
// logits[BT][V] = h_all[BT][2U] @ Wd + bd.  grid: BT/128, block 256.
// A + 6 B fragments register double-buffered.
__global__ void k_logits(const __bf16* __restrict__ h_all, const __bf16* __restrict__ Wdt,
                         const float* __restrict__ bd, float* __restrict__ out) {
    const int wave = threadIdx.x >> 5;
    const int m0   = blockIdx.x * 128 + wave * 16;
    const __bf16* abase = h_all + (size_t)m0 * (2 * U);

    v8f acc[6] = {};
    v16bf ac = load_frag_a(abase, 2 * U);
    v16bf bc[6];
#pragma unroll
    for (int j = 0; j < 6; ++j)
        bc[j] = load_frag_bt(Wdt + (size_t)(j * 16) * (2 * U), 2 * U);

    for (int k = 0; k < 2 * U; k += 32) {
        v16bf an = ac;
        v16bf bn[6];
#pragma unroll
        for (int j = 0; j < 6; ++j) bn[j] = bc[j];
        if (k + 32 < 2 * U) {
            an = load_frag_a(abase + k + 32, 2 * U);
#pragma unroll
            for (int j = 0; j < 6; ++j)
                bn[j] = load_frag_bt(Wdt + (size_t)(j * 16) * (2 * U) + k + 32, 2 * U);
        }
#pragma unroll
        for (int j = 0; j < 6; ++j)
            acc[j] = wmma_bf16(ac, bc[j], acc[j]);
        ac = an;
#pragma unroll
        for (int j = 0; j < 6; ++j) bc[j] = bn[j];
    }
#pragma unroll
    for (int j = 0; j < 6; ++j)
        store_c_bias(out, V, m0, j * 16, bd, acc[j]);
}

// ---------------- host side ----------------
extern "C" void kernel_launch(void* const* d_in, const int* in_sizes, int n_in,
                              void* d_out, int out_size, void* d_ws, size_t ws_size,
                              hipStream_t stream) {
    (void)in_sizes; (void)n_in; (void)out_size; (void)ws_size;
    const int*   tokens = (const int*)  d_in[0];
    const float* emb    = (const float*)d_in[1];
    const float* Wf     = (const float*)d_in[2];
    const float* Uf     = (const float*)d_in[3];
    const float* bfp    = (const float*)d_in[4];
    const float* Wb     = (const float*)d_in[5];
    const float* Ub     = (const float*)d_in[6];
    const float* bbp    = (const float*)d_in[7];
    const float* Wd     = (const float*)d_in[8];
    const float* bd     = (const float*)d_in[9];
    const float* h0f    = (const float*)d_in[10];
    const float* h0b    = (const float*)d_in[11];
    float* out = (float*)d_out;

    char* ws = (char*)d_ws;
    size_t off = 0;
    auto take = [&](size_t bytes) -> char* {
        char* p = ws + off;
        off = (off + bytes + 255) & ~(size_t)255;
        return p;
    };
    __bf16* Uft  = (__bf16*)take((size_t)U3 * U * 2);
    __bf16* Ubt  = (__bf16*)take((size_t)U3 * U * 2);
    __bf16* Wft  = (__bf16*)take((size_t)U3 * E * 2);
    __bf16* Wbt  = (__bf16*)take((size_t)U3 * E * 2);
    __bf16* Wdt  = (__bf16*)take((size_t)V * 2 * U * 2);
    __bf16* xbf  = (__bf16*)take((size_t)BT * E * 2);
    float*  xpF  = (float*) take((size_t)BT * U3 * 4);
    float*  xpB  = (float*) take((size_t)BT * U3 * 4);
    __bf16* hall = (__bf16*)take((size_t)BT * 2 * U * 2);
    __bf16* hb0  = (__bf16*)take((size_t)2 * B * U * 2);
    __bf16* hb1  = (__bf16*)take((size_t)2 * B * U * 2);

    // 1) pack weights (transpose + bf16)
    {
        int n;
        n = U * U3;    k_trans_bf16<<<(n + 255) / 256, 256, 0, stream>>>(Uf, Uft, U, U3);
        n = U * U3;    k_trans_bf16<<<(n + 255) / 256, 256, 0, stream>>>(Ub, Ubt, U, U3);
        n = E * U3;    k_trans_bf16<<<(n + 255) / 256, 256, 0, stream>>>(Wf, Wft, E, U3);
        n = E * U3;    k_trans_bf16<<<(n + 255) / 256, 256, 0, stream>>>(Wb, Wbt, E, U3);
        n = 2 * U * V; k_trans_bf16<<<(n + 255) / 256, 256, 0, stream>>>(Wd, Wdt, 2 * U, V);
    }
    // 2) embedding gather -> bf16
    {
        int n = BT * (E / 4);
        k_embed<<<(n + 255) / 256, 256, 0, stream>>>(tokens, emb, xbf);
    }
    // 3) initial hidden state -> bf16 ping buffer
    {
        int n = 2 * B * U;
        k_inith<<<(n + 255) / 256, 256, 0, stream>>>(h0f, h0b, hb0);
    }
    // 4) input projections for both directions (backward time-reversed)
    {
        dim3 grid(BT / 64, U3 / 128, 2);
        k_xproj<<<grid, 256, 0, stream>>>(xbf, Wft, Wbt, bfp, bbp, xpF, xpB);
    }
    // 5) recurrent scan: 512 sequential fused GEMM+gate steps (both dirs)
    {
        const float* bfr = bfp + U3;   // recurrent bias rows b[1]
        const float* bbr = bbp + U3;
        float* out_h = out + (size_t)BT * V;   // hf then hb after logits
        dim3 grid(U / 32, 2);
        for (int t = 0; t < T; ++t) {
            __bf16* cur = (t & 1) ? hb1 : hb0;
            __bf16* nxt = (t & 1) ? hb0 : hb1;
            k_gru_step<<<grid, 256, 0, stream>>>(t, cur, nxt, Uft, Ubt, bfr, bbr,
                                                 xpF, xpB, hall, out_h);
        }
    }
    // 6) decoder GEMM -> logits
    {
        dim3 grid(BT / 128);
        k_logits<<<grid, 256, 0, stream>>>(hall, Wdt, bd, out);
    }
}
